// FlaxPerformerModule_5179730559488
// MI455X (gfx1250) — matrix operand order repaired and checked by reading, not compiled, not run on hardware
//
#include <hip/hip_runtime.h>
#include <math.h>

// ---------------- model constants ----------------
#define BATCH   2
#define SEQ     2048
#define HID     768
#define NHEADS  12
#define HDIM    64
#define LAYERS  6
#define FFI     3072
#define MFEAT   256
#define ROWS    (BATCH * SEQ)          // 4096
#define QROWS   (BATCH * SEQ * NHEADS) // 49152

#define NEPS    1e-6f
#define DN      0.3535533905932738f    // 64^(-1/4)
#define RATIO   0.0625f                // 1/sqrt(256)

// ---------------- WMMA types ----------------
typedef __attribute__((ext_vector_type(16))) __bf16 v16bf;
typedef __attribute__((ext_vector_type(8)))  __bf16 v8bf;
typedef __attribute__((ext_vector_type(8)))  float  v8f;

// ---------------- shared-mem reductions (256 threads) ----------------
__device__ __forceinline__ float block_reduce_sum(float v, float* sh) {
    int tid = threadIdx.x;
    sh[tid] = v; __syncthreads();
    for (int off = 128; off > 0; off >>= 1) {
        if (tid < off) sh[tid] += sh[tid + off];
        __syncthreads();
    }
    float r = sh[0]; __syncthreads();
    return r;
}
__device__ __forceinline__ float block_reduce_max(float v, float* sh) {
    int tid = threadIdx.x;
    sh[tid] = v; __syncthreads();
    for (int off = 128; off > 0; off >>= 1) {
        if (tid < off) sh[tid] = fmaxf(sh[tid], sh[tid + off]);
        __syncthreads();
    }
    float r = sh[0]; __syncthreads();
    return r;
}

// ---------------- generic batched bf16-WMMA GEMM ----------------
// C = epilogue(alpha * A @ B + bias). Block tile 64x64, 128 threads
// (4 wave32s, each a 32x32 macro-tile). K is staged 64 per barrier round:
// every wave issues 8x v_wmma_f32_16x16x32_bf16 between barriers.
// LDS holds A as [m][k] and B transposed as [n][k], both padded, so every
// WMMA fragment is a contiguous 16B run -> ds_load_b128, no repacking.
#define PITCH 72   // halves per LDS row: 64 data + 8 pad

__device__ __forceinline__ v8bf cvt8(const float* s, float alpha) {
    v8bf r;
    #pragma unroll
    for (int i = 0; i < 8; ++i) r[i] = (__bf16)(alpha * s[i]);
    return r;
}

template<int TRANSA, int TRANSB, int GELU>
__launch_bounds__(128)
__global__ void gemm_wmma(const float* __restrict__ A,
                          const float* __restrict__ Bm,
                          const float* __restrict__ bias,
                          float* __restrict__ C,
                          int Kd, int lda, int ldb, int ldc,
                          float alpha, int batchInner,
                          int aO, int aI, int bO, int bI, int cO, int cI)
{
    __shared__ __bf16 sA[64 * PITCH];
    __shared__ __bf16 sB[64 * PITCH];

    int z  = blockIdx.z;
    int zo = z / batchInner, zi = z - zo * batchInner;
    A  += zo * aO + zi * aI;
    Bm += zo * bO + zi * bI;
    C  += zo * cO + zi * cI;

    const int m0  = blockIdx.x * 64;
    const int n0  = blockIdx.y * 64;
    const int tid = threadIdx.x;
    const int lane = tid & 31, wave = tid >> 5;
    const int wm = (wave >> 1) * 32, wn = (wave & 1) * 32;
    const int lh = lane >> 4;          // half-wave select
    const int hrow = lane & 15;

    // per-thread staging coordinates + hoisted global base pointers
    // (each thread moves 32 consecutive floats of A and of B per K round)
    int am, ak, aAdv;
    if (TRANSA == 0) { am = tid >> 1; ak = (tid & 1) * 32; aAdv = 64; }
    else             { ak = tid >> 1; am = (tid & 1) * 32; aAdv = 64 * lda; }
    const float* aSrc = TRANSA ? (A + ak * lda + m0 + am)
                               : (A + (m0 + am) * lda + ak);
    int bn, bk, bAdv;
    if (TRANSB == 0) { bk = tid >> 1; bn = (tid & 1) * 32; bAdv = 64 * ldb; }
    else             { bn = tid >> 1; bk = (tid & 1) * 32; bAdv = 64; }
    const float* bSrc = TRANSB ? (Bm + (n0 + bn) * ldb + bk)
                               : (Bm + bk * ldb + n0 + bn);

    v8f acc[2][2] = {};
    const int kb0 = lh * 16;           // A frag K offset
    const int koff = lh * 16;          // B frag K offset

    for (int k0 = 0; k0 < Kd; k0 += 64) {
        // ---- stage A tile (64x64 -> sA[m][k]) ----
        #pragma unroll
        for (int c = 0; c < 2; ++c) {
            float buf[16];
            const float4* s4 = (const float4*)(aSrc + 16 * c);
            #pragma unroll
            for (int i = 0; i < 4; ++i) *(float4*)&buf[4 * i] = s4[i];
            if (TRANSA == 0) {
                *(v8bf*)&sA[am * PITCH + ak + 16 * c]     = cvt8(&buf[0], alpha);
                *(v8bf*)&sA[am * PITCH + ak + 16 * c + 8] = cvt8(&buf[8], alpha);
            } else {
                #pragma unroll
                for (int i = 0; i < 16; ++i)
                    sA[(am + 16 * c + i) * PITCH + ak] = (__bf16)(alpha * buf[i]);
            }
        }
        // ---- stage B tile (64x64 -> sB[n][k], transposed) ----
        #pragma unroll
        for (int c = 0; c < 2; ++c) {
            float buf[16];
            const float4* s4 = (const float4*)(bSrc + 16 * c);
            #pragma unroll
            for (int i = 0; i < 4; ++i) *(float4*)&buf[4 * i] = s4[i];
            if (TRANSB == 0) {
                #pragma unroll
                for (int i = 0; i < 16; ++i)
                    sB[(bn + 16 * c + i) * PITCH + bk] = (__bf16)buf[i];
            } else {
                *(v8bf*)&sB[bn * PITCH + bk + 16 * c]     = cvt8(&buf[0], 1.f);
                *(v8bf*)&sB[bn * PITCH + bk + 16 * c + 8] = cvt8(&buf[8], 1.f);
            }
        }
        aSrc += aAdv;
        bSrc += bAdv;
        if (k0 + 64 < Kd) {            // prefetch next K tile
            __builtin_prefetch(aSrc, 0, 1);
            __builtin_prefetch(bSrc, 0, 1);
        }
        __syncthreads();

        // ---- 8 WMMAs per wave per barrier round (2M x 2N x 2K-halves) ----
        #pragma unroll
        for (int h = 0; h < 2; ++h) {
            v16bf afrag[2], bfrag[2];
            #pragma unroll
            for (int t = 0; t < 2; ++t) {
                const __bf16* As = &sA[(wm + t * 16 + hrow) * PITCH + h * 32 + kb0];
                v8bf a0 = *(const v8bf*)As;
                v8bf a1 = *(const v8bf*)(As + 8);
                afrag[t] = __builtin_shufflevector(a0, a1, 0, 1, 2, 3, 4, 5, 6, 7,
                                                   8, 9, 10, 11, 12, 13, 14, 15);
            }
            #pragma unroll
            for (int t = 0; t < 2; ++t) {
                const __bf16* Bs = &sB[(wn + t * 16 + hrow) * PITCH + h * 32 + koff];
                v8bf b0 = *(const v8bf*)Bs;
                v8bf b1 = *(const v8bf*)(Bs + 8);
                bfrag[t] = __builtin_shufflevector(b0, b1, 0, 1, 2, 3, 4, 5, 6, 7,
                                                   8, 9, 10, 11, 12, 13, 14, 15);
            }
            #pragma unroll
            for (int im = 0; im < 2; ++im)
                #pragma unroll
                for (int in = 0; in < 2; ++in)
                    acc[im][in] = __builtin_amdgcn_wmma_f32_16x16x32_bf16(
                        false, afrag[im], false, bfrag[in],
                        (short)0, acc[im][in], false, false);
        }
        __syncthreads();
    }

    // ---- epilogue: + bias, optional tanh-GELU ----
    #pragma unroll
    for (int im = 0; im < 2; ++im)
        #pragma unroll
        for (int in = 0; in < 2; ++in) {
            int col = n0 + wn + in * 16 + hrow;
            float bv = bias ? bias[col] : 0.f;
            #pragma unroll
            for (int r = 0; r < 8; ++r) {
                int row = m0 + wm + im * 16 + lh * 8 + r;
                float v = acc[im][in][r] + bv;
                if (GELU) {
                    float x3 = v * v * v;
                    v = 0.5f * v * (1.f + tanhf(0.7978845608028654f * (v + 0.044715f * x3)));
                }
                C[row * ldc + col] = v;
            }
        }
}

// ---------------- embeddings ----------------
__global__ void embed_kernel(const int* __restrict__ ids,
                             const int* __restrict__ tt,
                             const int* __restrict__ pos,
                             const float* __restrict__ wemb,
                             const float* __restrict__ pemb,
                             const float* __restrict__ temb,
                             float* __restrict__ x)
{
    int row = blockIdx.x;
    int id = ids[row], t = tt[row], p = pos[row];
    const float* w  = wemb + (long long)id * HID;
    const float* pe = pemb + (long long)p * HID;
    const float* te = temb + (long long)t * HID;
    for (int h = threadIdx.x; h < HID; h += blockDim.x)
        x[row * HID + h] = w[h] + pe[h] + te[h];
}

// ---------------- layernorm (y = LN(in1 (+ in2)) * g + b) ----------------
__global__ void layernorm_kernel(const float* __restrict__ in1,
                                 const float* __restrict__ in2,
                                 const float* __restrict__ g,
                                 const float* __restrict__ b,
                                 float* __restrict__ out)
{
    __shared__ float sh[256];
    int row = blockIdx.x, tid = threadIdx.x;
    const float* p1 = in1 + row * HID;
    const float* p2 = in2 ? in2 + row * HID : nullptr;
    float vals[3]; float s = 0.f, ss = 0.f;
    #pragma unroll
    for (int i = 0; i < 3; ++i) {
        int h = tid + i * 256;
        float v = p1[h] + (p2 ? p2[h] : 0.f);
        vals[i] = v; s += v; ss += v * v;
    }
    float mean = block_reduce_sum(s, sh) * (1.f / HID);
    float msq  = block_reduce_sum(ss, sh) * (1.f / HID);
    float rstd = rsqrtf(msq - mean * mean + NEPS);
    #pragma unroll
    for (int i = 0; i < 3; ++i) {
        int h = tid + i * 256;
        out[row * HID + h] = (vals[i] - mean) * rstd * g[h] + b[h];
    }
}

// ---------------- per-(b,head) global key max ----------------
__global__ void kmax_kernel(const float* __restrict__ kp, float* __restrict__ kmax)
{
    __shared__ float sh[256];
    int bn = blockIdx.x;
    int b = bn / NHEADS, n = bn - b * NHEADS;
    float m = -1e30f;
    for (int e = threadIdx.x; e < SEQ * MFEAT; e += 256) {
        int s = e >> 8, mm = e & 255;
        m = fmaxf(m, kp[(((b * SEQ + s) * NHEADS + n) * MFEAT) + mm]);
    }
    float r = block_reduce_max(m, sh);
    if (threadIdx.x == 0) kmax[bn] = r;
}

// ---------------- q' transform (in-place on qp) ----------------
__global__ void qprime_kernel(float* __restrict__ qp, const float* __restrict__ q)
{
    __shared__ float sh[256];
    int row = blockIdx.x;
    int tid = threadIdx.x;
    float v = qp[row * MFEAT + tid];
    float rmax = block_reduce_max(v, sh);
    float qv = (tid < HDIM) ? q[row * HDIM + tid] * DN : 0.f;
    float qd = 0.5f * block_reduce_sum(qv * qv, sh);
    qp[row * MFEAT + tid] = RATIO * (expf(v - qd - rmax) + NEPS);
}

// ---------------- k' transform (in-place on kp, masked) ----------------
__global__ void kprime_kernel(float* __restrict__ kp, const float* __restrict__ k,
                              const float* __restrict__ kmax,
                              const int* __restrict__ mask)
{
    __shared__ float sh[256];
    int row = blockIdx.x;                          // (b*SEQ+s)*NHEADS + n
    int b = row / (SEQ * NHEADS);
    int s = (row / NHEADS) % SEQ;
    int n = row % NHEADS;
    int tid = threadIdx.x;
    float v = kp[row * MFEAT + tid];
    float kv = (tid < HDIM) ? k[row * HDIM + tid] * DN : 0.f;
    float kd = 0.5f * block_reduce_sum(kv * kv, sh);
    float km = kmax[b * NHEADS + n];
    float mv = (float)mask[b * SEQ + s];
    kp[row * MFEAT + tid] = RATIO * (expf(v - kd - km) + NEPS) * mv;
}

// ---------------- ksum over sequence ----------------
__global__ void ksum_kernel(const float* __restrict__ kp, float* __restrict__ ksum)
{
    int bn = blockIdx.x;
    int b = bn / NHEADS, n = bn - b * NHEADS;
    int m = threadIdx.x;                           // MFEAT == 256
    float acc = 0.f;
    for (int s = 0; s < SEQ; ++s)
        acc += kp[(((b * SEQ + s) * NHEADS + n) * MFEAT) + m];
    ksum[bn * MFEAT + m] = acc;
}

// ---------------- denominator + divide (att = num / den) ----------------
__global__ void den_div_kernel(const float* __restrict__ qp,
                               const float* __restrict__ ksum,
                               float* __restrict__ att)
{
    __shared__ float sh[256];
    int row = blockIdx.x;                          // (b*SEQ+s)*NHEADS + n
    int b = row / (SEQ * NHEADS);
    int n = row % NHEADS;
    int tid = threadIdx.x;
    float p = qp[row * MFEAT + tid] * ksum[(b * NHEADS + n) * MFEAT + tid];
    float den = block_reduce_sum(p, sh);
    float inv = 1.f / den;
    if (tid < HDIM) att[row * HDIM + tid] *= inv;
}

// ---------------- host-side launcher ----------------
struct GemmArgs {
    const float* A; const float* B; const float* bias; float* C;
    int Md, Nd, Kd, lda, ldb, ldc;
    float alpha;
    int batches, batchInner;
    int aO, aI, bO, bI, cO, cI;
};

template<int TA, int TB, int G>
static inline void launch_gemm_t(hipStream_t st, const GemmArgs& a)
{
    dim3 grid(a.Md / 64, a.Nd / 64, a.batches);
    gemm_wmma<TA, TB, G><<<grid, 128, 0, st>>>(a.A, a.B, a.bias, a.C,
                                               a.Kd, a.lda, a.ldb, a.ldc,
                                               a.alpha, a.batchInner,
                                               a.aO, a.aI, a.bO, a.bI, a.cO, a.cI);
}

extern "C" void kernel_launch(void* const* d_in, const int* in_sizes, int n_in,
                              void* d_out, int out_size, void* d_ws, size_t ws_size,
                              hipStream_t stream)
{
    (void)in_sizes; (void)n_in; (void)out_size; (void)ws_size;

    const int*   input_ids      = (const int*)  d_in[0];
    const int*   token_type_ids = (const int*)  d_in[1];
    const int*   position_ids   = (const int*)  d_in[2];
    const int*   attention_mask = (const int*)  d_in[3];
    const float* word_emb       = (const float*)d_in[4];
    const float* pos_emb        = (const float*)d_in[5];
    const float* type_emb       = (const float*)d_in[6];
    const float* emb_ln_g       = (const float*)d_in[7];
    const float* emb_ln_b       = (const float*)d_in[8];
    const float* wq             = (const float*)d_in[9];
    const float* bq             = (const float*)d_in[10];
    const float* wk             = (const float*)d_in[11];
    const float* bk             = (const float*)d_in[12];
    const float* wv             = (const float*)d_in[13];
    const float* bv             = (const float*)d_in[14];
    const float* wo             = (const float*)d_in[15];
    const float* bo             = (const float*)d_in[16];
    const float* ln1_g          = (const float*)d_in[17];
    const float* ln1_b          = (const float*)d_in[18];
    const float* wi             = (const float*)d_in[19];
    const float* bi             = (const float*)d_in[20];
    const float* wo2            = (const float*)d_in[21];
    const float* bo2            = (const float*)d_in[22];
    const float* ln2_g          = (const float*)d_in[23];
    const float* ln2_b          = (const float*)d_in[24];
    const float* proj           = (const float*)d_in[25];
    float* out = (float*)d_out;

    // workspace layout
    float* p = (float*)d_ws;
    const long long SZ_X  = (long long)ROWS * HID;
    const long long SZ_QP = (long long)QROWS * MFEAT;
    float* xbuf    = p; p += SZ_X;
    float* abuf    = p; p += SZ_X;
    float* qbuf    = p; p += SZ_X;
    float* kbuf    = p; p += SZ_X;
    float* vbuf    = p; p += SZ_X;
    float* attbuf  = p; p += SZ_X;
    float* tmpbuf  = p; p += SZ_X;
    float* interbuf = p; p += (long long)ROWS * FFI;
    float* qpbuf   = p; p += SZ_QP;
    float* kpbuf   = p; p += SZ_QP;
    float* kvbuf   = p; p += (long long)BATCH * NHEADS * MFEAT * HDIM;
    float* ksumbuf = p; p += (long long)BATCH * NHEADS * MFEAT;
    float* kmaxbuf = p; p += BATCH * NHEADS;

    // ---- embeddings + LN ----
    embed_kernel<<<ROWS, 256, 0, stream>>>(input_ids, token_type_ids, position_ids,
                                           word_emb, pos_emb, type_emb, tmpbuf);
    layernorm_kernel<<<ROWS, 256, 0, stream>>>(tmpbuf, nullptr, emb_ln_g, emb_ln_b, xbuf);

    const long long WQKV = (long long)HID * NHEADS * HDIM;

    for (int l = 0; l < LAYERS; ++l) {
        const float* wq_l = wq + l * WQKV;
        const float* wk_l = wk + l * WQKV;
        const float* wv_l = wv + l * WQKV;
        const float* wo_l = wo + l * WQKV;
        const float* bq_l = bq + l * HID;
        const float* bk_l = bk + l * HID;
        const float* bv_l = bv + l * HID;
        const float* bo_l = bo + l * HID;
        const float* wi_l = wi + (long long)l * HID * FFI;
        const float* bi_l = bi + l * FFI;
        const float* wo2_l = wo2 + (long long)l * FFI * HID;
        const float* bo2_l = bo2 + l * HID;
        const float* proj_l = proj + l * MFEAT * HDIM;

        // Q,K,V projections: [4096,768] @ [768,768] + bias
        GemmArgs g{};
        g.Md = ROWS; g.Nd = HID; g.Kd = HID; g.lda = HID; g.ldb = HID; g.ldc = HID;
        g.alpha = 1.f; g.batches = 1; g.batchInner = 1;
        g.A = xbuf; g.B = wq_l; g.bias = bq_l; g.C = qbuf; launch_gemm_t<0,0,0>(stream, g);
        g.B = wk_l; g.bias = bk_l; g.C = kbuf; launch_gemm_t<0,0,0>(stream, g);
        g.B = wv_l; g.bias = bv_l; g.C = vbuf; launch_gemm_t<0,0,0>(stream, g);

        // random-feature projections: qp = dn * q @ proj^T   [49152,64]@[64,256]
        GemmArgs g2{};
        g2.Md = QROWS; g2.Nd = MFEAT; g2.Kd = HDIM;
        g2.lda = HDIM; g2.ldb = HDIM; g2.ldc = MFEAT;
        g2.alpha = DN; g2.batches = 1; g2.batchInner = 1;
        g2.A = qbuf; g2.B = proj_l; g2.bias = nullptr; g2.C = qpbuf;
        launch_gemm_t<0,1,0>(stream, g2);
        g2.A = kbuf; g2.C = kpbuf;
        launch_gemm_t<0,1,0>(stream, g2);

        // stabilizers + exponentials
        kmax_kernel<<<BATCH * NHEADS, 256, 0, stream>>>(kpbuf, kmaxbuf);
        kprime_kernel<<<QROWS, 256, 0, stream>>>(kpbuf, kbuf, kmaxbuf, attention_mask);
        qprime_kernel<<<QROWS, 256, 0, stream>>>(qpbuf, qbuf);
        ksum_kernel<<<BATCH * NHEADS, 256, 0, stream>>>(kpbuf, ksumbuf);

        // kv[b,n] = k'^T @ v : [256,2048]@[2048,64], batched over (b,n)
        GemmArgs g3{};
        g3.Md = MFEAT; g3.Nd = HDIM; g3.Kd = SEQ;
        g3.lda = NHEADS * MFEAT; g3.ldb = NHEADS * HDIM; g3.ldc = HDIM;
        g3.alpha = 1.f; g3.batches = BATCH * NHEADS; g3.batchInner = NHEADS;
        g3.aO = SEQ * NHEADS * MFEAT; g3.aI = MFEAT;
        g3.bO = SEQ * NHEADS * HDIM;  g3.bI = HDIM;
        g3.cO = NHEADS * MFEAT * HDIM; g3.cI = MFEAT * HDIM;
        g3.A = kpbuf; g3.B = vbuf; g3.bias = nullptr; g3.C = kvbuf;
        launch_gemm_t<1,0,0>(stream, g3);

        // num[b,n] = q' @ kv : [2048,256]@[256,64], batched over (b,n)
        GemmArgs g4{};
        g4.Md = SEQ; g4.Nd = HDIM; g4.Kd = MFEAT;
        g4.lda = NHEADS * MFEAT; g4.ldb = HDIM; g4.ldc = NHEADS * HDIM;
        g4.alpha = 1.f; g4.batches = BATCH * NHEADS; g4.batchInner = NHEADS;
        g4.aO = SEQ * NHEADS * MFEAT;  g4.aI = MFEAT;
        g4.bO = NHEADS * MFEAT * HDIM; g4.bI = MFEAT * HDIM;
        g4.cO = SEQ * NHEADS * HDIM;   g4.cI = HDIM;
        g4.A = qpbuf; g4.B = kvbuf; g4.bias = nullptr; g4.C = attbuf;
        launch_gemm_t<0,0,0>(stream, g4);

        // denominator + divide
        den_div_kernel<<<QROWS, 256, 0, stream>>>(qpbuf, ksumbuf, attbuf);

        // attention output projection
        GemmArgs g5{};
        g5.Md = ROWS; g5.Nd = HID; g5.Kd = HID; g5.lda = HID; g5.ldb = HID; g5.ldc = HID;
        g5.alpha = 1.f; g5.batches = 1; g5.batchInner = 1;
        g5.A = attbuf; g5.B = wo_l; g5.bias = bo_l; g5.C = tmpbuf;
        launch_gemm_t<0,0,0>(stream, g5);

        // a = LN(att_out + x)
        layernorm_kernel<<<ROWS, 256, 0, stream>>>(tmpbuf, xbuf,
                                                   ln1_g + l * HID, ln1_b + l * HID, abuf);
        // FFN1 (+ GELU)
        GemmArgs g6{};
        g6.Md = ROWS; g6.Nd = FFI; g6.Kd = HID; g6.lda = HID; g6.ldb = FFI; g6.ldc = FFI;
        g6.alpha = 1.f; g6.batches = 1; g6.batchInner = 1;
        g6.A = abuf; g6.B = wi_l; g6.bias = bi_l; g6.C = interbuf;
        launch_gemm_t<0,0,1>(stream, g6);
        // FFN2
        GemmArgs g7{};
        g7.Md = ROWS; g7.Nd = HID; g7.Kd = FFI; g7.lda = FFI; g7.ldb = HID; g7.ldc = HID;
        g7.alpha = 1.f; g7.batches = 1; g7.batchInner = 1;
        g7.A = interbuf; g7.B = wo2_l; g7.bias = bo2_l; g7.C = tmpbuf;
        launch_gemm_t<0,0,0>(stream, g7);

        // x = LN(ffn_out + a)
        float* dest = (l == LAYERS - 1) ? out : xbuf;
        layernorm_kernel<<<ROWS, 256, 0, stream>>>(tmpbuf, abuf,
                                                   ln2_g + l * HID, ln2_b + l * HID, dest);
    }
}
